// MOE_33492154974359
// MI455X (gfx1250) — matrix operand order repaired and checked by reading, not compiled, not run on hardware
//
#include <hip/hip_runtime.h>
#include <hip/hip_bf16.h>
#include <math.h>

// Problem constants (match the reference)
#define B_SZ  4096
#define N_EXP 32
#define IDIM  1024
#define ODIM  1024

// LDS x-tile: [hi/lo][32 rows][1024 + 8 pad] bf16. 16B row padding spreads the
// 16 lanes of a ds_load_b128 across all 64 LDS banks (row stride 2064B ≡ 16 mod 256).
#define XROWS 32
#define XSTR  1032

typedef __attribute__((ext_vector_type(16))) __bf16 v16bf;
typedef __attribute__((ext_vector_type(8)))  __bf16 v8bf;
typedef __attribute__((ext_vector_type(4)))  __bf16 v4bf;
typedef __attribute__((ext_vector_type(8)))  float  v8f;
typedef __attribute__((ext_vector_type(4)))  float  v4f;

// ---------------------------------------------------------------------------
// Split an f32 array into bf16 hi + bf16 residual (lo) arrays: f = hi + lo.
// ---------------------------------------------------------------------------
__global__ void moe_split_bf16x2(const float* __restrict__ src,
                                 __bf16* __restrict__ hi,
                                 __bf16* __restrict__ lo,
                                 long long n4) {
  long long i = (long long)blockIdx.x * blockDim.x + threadIdx.x;
  if (i >= n4) return;
  v4f f = *(((const v4f*)src) + i);
  v4bf h, l;
#pragma unroll
  for (int j = 0; j < 4; ++j) {
    __bf16 hh = (__bf16)f[j];
    h[j] = hh;
    l[j] = (__bf16)(f[j] - (float)hh);
  }
  *(((v4bf*)hi) + i) = h;
  *(((v4bf*)lo) + i) = l;
}

// ---------------------------------------------------------------------------
// Gate: g[b,n] = gelu_exact(x[b,:] . gate_w[n,:] + gate_b[n])
// ---------------------------------------------------------------------------
__global__ void moe_gate_kernel(const float* __restrict__ x,
                                const float* __restrict__ gw,
                                const float* __restrict__ gb,
                                float* __restrict__ g) {
  int idx = blockIdx.x * blockDim.x + threadIdx.x;   // 0 .. B*N-1
  int b = idx >> 5;
  int n = idx & 31;
  const v4f* xr = (const v4f*)(x + (size_t)b * IDIM);
  const v4f* wr = (const v4f*)(gw + (size_t)n * IDIM);
  float s = gb[n];
#pragma unroll 4
  for (int i = 0; i < IDIM / 4; ++i) {
    v4f xv = xr[i];
    v4f wv = wr[i];
    s += xv[0] * wv[0] + xv[1] * wv[1] + xv[2] * wv[2] + xv[3] * wv[3];
  }
  g[idx] = 0.5f * s * (1.0f + erff(s * 0.70710678118654752f));
}

// ---------------------------------------------------------------------------
// Fused MoE projection + gated combine.
// WG = 8 waves; WG covers 32 b-rows x 128 o-cols. Each wave owns TWO 16x16
// output tiles (b-tile 0 and 1) that share B fragments -> halves L2 weight
// refetch vs a 16-row tile. x tile staged bf16(hi/lo) into LDS via
// global_load_async_to_lds_b128 (ASYNCcnt), A fed from ds_load_b128.
// bf16x3 split (hi*hi + hi*lo + lo*hi) with f32 accumulation ~ fp32 accuracy.
// ---------------------------------------------------------------------------
__global__ void __launch_bounds__(256) moe_wmma_kernel(
    const __bf16* __restrict__ Xhi, const __bf16* __restrict__ Xlo,
    const __bf16* __restrict__ Whi, const __bf16* __restrict__ Wlo,
    const float*  __restrict__ G,   float* __restrict__ Out) {
  __shared__ __bf16 sX[2][XROWS][XSTR];

  const int lane = threadIdx.x & 31;
  const int wave = threadIdx.x >> 5;
  const int col  = lane & 15;   // A row-in-tile / B column / output column
  const int hs   = lane >> 4;   // lane half select
  const int b0   = blockIdx.x * XROWS;
  const int o0   = blockIdx.y * 128 + wave * 16;

  // ---- Stage x tile (hi+lo) into LDS with async global->LDS copies --------
  // 2 mats * 32 rows * 128 cols of 16B = 8192 chunks; 32 per thread.
  {
    unsigned ldsbase = (unsigned)(unsigned long long)(const void*)&sX[0][0][0];
#pragma unroll
    for (int i = 0; i < 32; ++i) {
      int c   = (int)threadIdx.x + 256 * i;   // 0..8191
      int mat = c >> 12;
      int rem = c & 4095;
      int row = rem >> 7;                     // 0..31
      int c16 = rem & 127;                    // 16B column
      const __bf16* src = (mat ? Xlo : Xhi) + (size_t)(b0 + row) * IDIM + c16 * 8;
      unsigned lds = ldsbase +
                     (unsigned)(mat * (XROWS * XSTR * 2) + row * (XSTR * 2) + c16 * 16);
      asm volatile("global_load_async_to_lds_b128 %0, %1, off"
                   :: "v"(lds), "v"((unsigned long long)src)
                   : "memory");
    }
    asm volatile("s_wait_asynccnt 0" ::: "memory");
    __syncthreads();
  }

  // A fragment base pointers (LDS): tile0 rows 0-15, tile1 rows 16-31.
  const __bf16* sxh0 = &sX[0][col][8 * hs];
  const __bf16* sxh1 = &sX[0][16 + col][8 * hs];
  const __bf16* sxl0 = &sX[1][col][8 * hs];
  const __bf16* sxl1 = &sX[1][16 + col][8 * hs];

  const size_t woff = (size_t)(o0 + col) * IDIM + 8 * hs;

  v8f outacc0 = {};
  v8f outacc1 = {};
#pragma unroll 1
  for (int n = 0; n < N_EXP; ++n) {
    const __bf16* wh = Whi + (size_t)n * ((size_t)ODIM * IDIM) + woff;
    const __bf16* wl = Wlo + (size_t)n * ((size_t)ODIM * IDIM) + woff;
    v8f acc0 = {};
    v8f acc1 = {};
#pragma unroll 2
    for (int k0 = 0; k0 < IDIM; k0 += 32) {
      // B fragments (global, K-contiguous)
      v8bf bh0 = *(const v8bf*)(wh + k0);
      v8bf bh1 = *(const v8bf*)(wh + k0 + 16);
      v8bf bl0 = *(const v8bf*)(wl + k0);
      v8bf bl1 = *(const v8bf*)(wl + k0 + 16);
      v16bf Bhi = __builtin_shufflevector(bh0, bh1, 0,1,2,3,4,5,6,7,8,9,10,11,12,13,14,15);
      v16bf Blo = __builtin_shufflevector(bl0, bl1, 0,1,2,3,4,5,6,7,8,9,10,11,12,13,14,15);
      // A fragments (LDS)
      v8bf a0h0 = *(const v8bf*)(sxh0 + k0);
      v8bf a0h1 = *(const v8bf*)(sxh0 + k0 + 16);
      v8bf a0l0 = *(const v8bf*)(sxl0 + k0);
      v8bf a0l1 = *(const v8bf*)(sxl0 + k0 + 16);
      v8bf a1h0 = *(const v8bf*)(sxh1 + k0);
      v8bf a1h1 = *(const v8bf*)(sxh1 + k0 + 16);
      v8bf a1l0 = *(const v8bf*)(sxl1 + k0);
      v8bf a1l1 = *(const v8bf*)(sxl1 + k0 + 16);
      v16bf A0hi = __builtin_shufflevector(a0h0, a0h1, 0,1,2,3,4,5,6,7,8,9,10,11,12,13,14,15);
      v16bf A0lo = __builtin_shufflevector(a0l0, a0l1, 0,1,2,3,4,5,6,7,8,9,10,11,12,13,14,15);
      v16bf A1hi = __builtin_shufflevector(a1h0, a1h1, 0,1,2,3,4,5,6,7,8,9,10,11,12,13,14,15);
      v16bf A1lo = __builtin_shufflevector(a1l0, a1l1, 0,1,2,3,4,5,6,7,8,9,10,11,12,13,14,15);
      // D = A*B + C ; bf16x3 split for near-fp32 accuracy, both b-tiles.
      acc0 = __builtin_amdgcn_wmma_f32_16x16x32_bf16(false, A0hi, false, Bhi,
                                                     (short)0, acc0, false, false);
      acc0 = __builtin_amdgcn_wmma_f32_16x16x32_bf16(false, A0hi, false, Blo,
                                                     (short)0, acc0, false, false);
      acc0 = __builtin_amdgcn_wmma_f32_16x16x32_bf16(false, A0lo, false, Bhi,
                                                     (short)0, acc0, false, false);
      acc1 = __builtin_amdgcn_wmma_f32_16x16x32_bf16(false, A1hi, false, Bhi,
                                                     (short)0, acc1, false, false);
      acc1 = __builtin_amdgcn_wmma_f32_16x16x32_bf16(false, A1hi, false, Blo,
                                                     (short)0, acc1, false, false);
      acc1 = __builtin_amdgcn_wmma_f32_16x16x32_bf16(false, A1lo, false, Bhi,
                                                     (short)0, acc1, false, false);
    }
    // Gated accumulate: element v of tile t sits at row b0 + 16*t + 8*hs + v
    const float* g0 = G + (size_t)(b0 + 8 * hs) * N_EXP + n;
    const float* g1 = G + (size_t)(b0 + 16 + 8 * hs) * N_EXP + n;
#pragma unroll
    for (int v = 0; v < 8; ++v) {
      outacc0[v] += g0[v * N_EXP] * acc0[v];
      outacc1[v] += g1[v * N_EXP] * acc1[v];
    }
  }

  float* o0p = Out + (size_t)(b0 + 8 * hs) * ODIM + o0 + col;
  float* o1p = Out + (size_t)(b0 + 16 + 8 * hs) * ODIM + o0 + col;
#pragma unroll
  for (int v = 0; v < 8; ++v) {
    o0p[(size_t)v * ODIM] = outacc0[v];
    o1p[(size_t)v * ODIM] = outacc1[v];
  }
}

// ---------------------------------------------------------------------------
// Launch
// ---------------------------------------------------------------------------
extern "C" void kernel_launch(void* const* d_in, const int* in_sizes, int n_in,
                              void* d_out, int out_size, void* d_ws, size_t ws_size,
                              hipStream_t stream) {
  const float* x  = (const float*)d_in[0];  // [B, IDIM]
  const float* w  = (const float*)d_in[1];  // [N, ODIM, IDIM]
  const float* gw = (const float*)d_in[2];  // [N, IDIM]
  const float* gb = (const float*)d_in[3];  // [N]
  float* out = (float*)d_out;               // [B, ODIM]

  // Workspace layout (bytes), all 16B-aligned:
  //   g    : B*N*4            =    524,288
  //   Xhi  : B*IDIM*2         =  8,388,608
  //   Xlo  : B*IDIM*2         =  8,388,608
  //   Whi  : N*ODIM*IDIM*2    = 67,108,864
  //   Wlo  : N*ODIM*IDIM*2    = 67,108,864   (total ~144.5 MB)
  char* ws = (char*)d_ws;
  float*  g   = (float*)ws;
  __bf16* Xhi = (__bf16*)(ws + 524288ULL);
  __bf16* Xlo = (__bf16*)(ws + 524288ULL + 8388608ULL);
  __bf16* Whi = (__bf16*)(ws + 524288ULL + 2ULL * 8388608ULL);
  __bf16* Wlo = (__bf16*)(ws + 524288ULL + 2ULL * 8388608ULL + 67108864ULL);

  // 1) Precision-split x and W into bf16 hi/lo
  {
    long long n4 = (long long)B_SZ * IDIM / 4;           // 1,048,576
    moe_split_bf16x2<<<(unsigned)((n4 + 255) / 256), 256, 0, stream>>>(x, Xhi, Xlo, n4);
  }
  {
    long long n4 = (long long)N_EXP * ODIM * IDIM / 4;   // 8,388,608
    moe_split_bf16x2<<<(unsigned)((n4 + 255) / 256), 256, 0, stream>>>(w, Whi, Wlo, n4);
  }

  // 2) Gate values g[b,n]
  moe_gate_kernel<<<(B_SZ * N_EXP) / 256, 256, 0, stream>>>(x, gw, gb, g);

  // 3) Fused per-expert GEMM + gated combine
  dim3 grid(B_SZ / XROWS, ODIM / 128);   // (128, 8)
  moe_wmma_kernel<<<grid, 256, 0, stream>>>(Xhi, Xlo, Whi, Wlo, g, out);
}
